// NoiseProjector_59837484368508
// MI455X (gfx1250) — matrix operand order repaired and checked by reading, not compiled, not run on hardware
//
#include <hip/hip_runtime.h>
#include <hip/hip_bf16.h>

// ---------------- CDNA5 WMMA types ----------------
typedef __bf16 bf16;
typedef bf16  v16bf __attribute__((ext_vector_type(16)));
typedef float v8f   __attribute__((ext_vector_type(8)));
typedef int   v4i   __attribute__((ext_vector_type(4)));

// ---------------- optional gfx1250 async-to-LDS path ----------------
#if defined(__has_builtin)
#  if __has_builtin(__builtin_amdgcn_global_load_async_to_lds_b128) && \
      __has_builtin(__builtin_amdgcn_s_wait_asynccnt)
#    define POLY_ASYNC 1
#  endif
#endif
#ifndef POLY_ASYNC
#  define POLY_ASYNC 0
#endif

#if POLY_ASYNC
typedef __attribute__((address_space(1))) v4i* as1p;   // global int4*
typedef __attribute__((address_space(3))) v4i* as3p;   // LDS int4*
#endif

// ======================================================================
// K1: conv1 3x3 s2 pad1 + bias + relu : x(64,3,224,224) -> h1b(64,32,112,112) bf16
// ======================================================================
__global__ __launch_bounds__(256) void conv1_kernel(
    const float* __restrict__ x, const float* __restrict__ w1,
    const float* __restrict__ b1, bf16* __restrict__ h1b)
{
    const int p  = blockIdx.x * 256 + threadIdx.x;     // 0..12543
    if (p >= 112 * 112) return;
    const int oc = blockIdx.y;
    const int b  = blockIdx.z;
    const int oy = p / 112, ox = p % 112;

    float acc = b1[oc];
#pragma unroll
    for (int ic = 0; ic < 3; ++ic) {
#pragma unroll
        for (int ky = 0; ky < 3; ++ky) {
            const int iy = 2 * oy + ky - 1;
            if (iy < 0 || iy >= 224) continue;
#pragma unroll
            for (int kx = 0; kx < 3; ++kx) {
                const int ix = 2 * ox + kx - 1;
                if (ix < 0 || ix >= 224) continue;
                acc += x[((b * 3 + ic) * 224 + iy) * 224 + ix] *
                       w1[((oc * 3 + ic) * 3 + ky) * 3 + kx];
            }
        }
    }
    h1b[((b * 32 + oc) * 112 + oy) * 112 + ox] = (bf16)fmaxf(acc, 0.0f);
}

// ======================================================================
// K2: conv2 as im2col WMMA GEMM, fused bias+relu+global-sum pool.
// Block = 128 thr (4 waves) = one (batch, 8x8 output-pixel tile).
// GEMM: M=64 pixels x N=64 oc x K=288 (k = ic*9 + ky*3 + kx == w2 layout).
// A-fragment gather uses a half-wave branch so every k decode is a
// compile-time constant -> pure ds_load_u16 with immediate offsets.
// ======================================================================
__device__ __forceinline__ int im2col_off(int k) {   // consteval under unroll
    const int ic = k / 9, kk = k % 9;
    return ic * 289 + (kk / 3) * 17 + (kk % 3);
}

__global__ __launch_bounds__(128) void conv2_pool_kernel(
    const bf16* __restrict__ h1b, const float* __restrict__ w2,
    const float* __restrict__ b2, float* __restrict__ featpart)
{
    __shared__ __align__(16) bf16 sIn[32 * 289];   // [ic][17*17] input slice (padded)
    __shared__ __align__(16) bf16 sW[64 * 296];    // [oc][k], row stride 296
    __shared__ float sFeat[64];

    const int tid  = threadIdx.x;                  // 128
    const int lane = tid & 31, wv = tid >> 5;
    const int t    = blockIdx.x;                   // 0..48 (7x7 tiles of 8x8)
    const int b    = blockIdx.y;
    const int tY   = t / 7, tX = t % 7;
    const int iy0  = tY * 16 - 1, ix0 = tX * 16 - 1;

    if (tid < 64) sFeat[tid] = 0.0f;

    for (int i = tid; i < 32 * 289; i += 128) {
        const int ic = i / 289, j = i % 289;
        const int r = j / 17, c = j % 17;
        const int iy = iy0 + r, ix = ix0 + c;
        bf16 v = (bf16)0.0f;
        if (iy >= 0 && iy < 112 && ix >= 0 && ix < 112)
            v = h1b[((b * 32 + ic) * 112 + iy) * 112 + ix];
        sIn[i] = v;
    }
    for (int i = tid; i < 64 * 288; i += 128)
        sW[(i / 288) * 296 + (i % 288)] = (bf16)w2[i];
    __syncthreads();

    const int n     = lane & 15;
    const int koffB = (lane >> 4) << 3;   // B-frag: lanes0-15 K {0-7,16-23}, lanes16-31 K {8-15,24-31}
    const int p     = (wv << 4) | n;      // pixel 0..63 in 8x8 tile
    const int py    = p >> 3, px = p & 7;
    const bf16* inp = sIn + (34 * py + 2 * px);

    v8f acc[4] = {};
#pragma unroll
    for (int ch = 0; ch < 9; ++ch) {
        const int k0 = ch * 32;
        v16bf afr;
        if (lane < 16) {      // K runs {0-7, 16-23}: all offsets compile-time
#pragma unroll
            for (int e = 0; e < 16; ++e)
                afr[e] = inp[im2col_off(k0 + (e < 8 ? e : e + 8))];
        } else {              // K runs {8-15, 24-31}
#pragma unroll
            for (int e = 0; e < 16; ++e)
                afr[e] = inp[im2col_off(k0 + 8 + (e < 8 ? e : e + 8))];
        }
#pragma unroll
        for (int nt = 0; nt < 4; ++nt) {
            v16bf bfr;
            const bf16* bp = &sW[(nt * 16 + n) * 296 + k0 + koffB];
#pragma unroll
            for (int e = 0; e < 16; ++e) bfr[e] = bp[e < 8 ? e : e + 8];
            acc[nt] = __builtin_amdgcn_wmma_f32_16x16x32_bf16(
                false, afr, false, bfr, (short)0, acc[nt], false, false);
        }
    }

    // bias + relu + pool-sum (pixel identity irrelevant for the sum)
#pragma unroll
    for (int nt = 0; nt < 4; ++nt) {
        const int oc = nt * 16 + (lane & 15);
        const float bb = b2[oc];
#pragma unroll
        for (int v = 0; v < 8; ++v)
            atomicAdd(&sFeat[oc], fmaxf(acc[nt][v] + bb, 0.0f));
    }
    __syncthreads();
    if (tid < 64)
        featpart[(b * 49 + t) * 64 + tid] = sFeat[tid];
}

// ======================================================================
// K3: feat = sum(49 partials)/3136 ; fc = feat - rowmean(feat)
// ======================================================================
__global__ __launch_bounds__(64) void feat_fc_kernel(
    const float* __restrict__ featpart, float* __restrict__ feat,
    float* __restrict__ fc)
{
    const int b = blockIdx.x, t = threadIdx.x;   // 64 threads
    float s = 0.0f;
    for (int i = 0; i < 49; ++i) s += featpart[(b * 49 + i) * 64 + t];
    const float v = s * (1.0f / 3136.0f);
    feat[b * 64 + t] = v;

    __shared__ float red[64];
    red[t] = v;
    __syncthreads();
    for (int off = 32; off > 0; off >>= 1) {
        if (t < off) red[t] += red[t + off];
        __syncthreads();
    }
    fc[b * 64 + t] = v - red[0] * (1.0f / 64.0f);
}

// ======================================================================
// K4: polynomial-feature GEMM  C(64 x 1024) += A(64 x K) @ W^T
//   MODE 1: A[b,k]         = feat[b,k]                 (K = 64)
//   MODE 2: A[b,i*64+j]    = fc[b,i]*fc[b,j]           (K = 4096)
//   MODE 3: A[b,(ij)*64+k] = fc[b,i]*fc[b,j]*fc[b,k]   (K = 262144)
// A generated on the fly (rank-1). W streamed CB K-chunks (16 x CB*32 fp32)
// per stage, async-to-LDS double-buffered where builtins exist (MODE 3 uses
// a non-temporal hint: 1.07 GB single-pass stream >> 192 MB L2), else
// synchronous float4 copies. bf16 WMMA, fp32 accumulation, deterministic
// K-split partial tiles.
// ======================================================================
template <int MODE>
__global__ __launch_bounds__(128) void poly_gemm_kernel(
    const float* __restrict__ W, const float* __restrict__ F,
    float* __restrict__ partial, int chunksPerSplit, int slotBase)
{
    constexpr int CB    = (MODE == 1) ? 2 : 8;       // K-chunks per stage
    constexpr int WTSTR = CB * 32 + 4;               // fp32 row stride (16B aligned)
    constexpr int CPOL  = (MODE == 3) ? 1 : 0;       // TH_NT for the w3 stream
    __shared__ float fcs[64 * 64];
    __shared__ __align__(16) float wt[2][16 * WTSTR];

    const int tid   = threadIdx.x;              // 128
    const int lane  = tid & 31;
    const int wv    = tid >> 5;
    const int obase = blockIdx.x << 4;
    const int split = blockIdx.y;
    const long long Ktot = (MODE == 3) ? 262144LL : (MODE == 2 ? 4096LL : 64LL);

    for (int i = tid; i < 4096; i += 128) fcs[i] = F[i];

    const int n     = lane & 15;
    const int koffA = (lane >> 4) << 3;
    const int koffB = (lane >> 4) << 3;
    const int brow  = (wv << 4) | n;
    const float* frow = &fcs[brow * 64];

    const int c0 = split * chunksPerSplit;
    const int nB = chunksPerSplit / CB;
    const int ldr = tid >> 3, ldc = (tid & 7) << 2;

    // issue one stage (CB chunks = 16 x CB*32 fp32) into buffer `buf`
    auto issue = [&](int ib, int buf) {
        if (ib >= nB) return;
        const long long kk = (long long)(c0 + ib * CB) << 5;
        const float* g = W + (long long)(obase + ldr) * Ktot + kk + ldc;
        float* d = &wt[buf][ldr * WTSTR + ldc];
#pragma unroll
        for (int cc = 0; cc < CB; ++cc) {
#if POLY_ASYNC
            __builtin_amdgcn_global_load_async_to_lds_b128(
                (as1p)(g + cc * 32), (as3p)(d + cc * 32), 0, CPOL);
#else
            *(float4*)(d + cc * 32) = *(const float4*)(g + cc * 32);
#endif
        }
    };

    v8f acc = {};

    auto compute = [&](int ib, int buf) {
#pragma unroll
        for (int cc = 0; cc < CB; ++cc) {
            const long long k0 = (long long)(c0 + ib * CB + cc) << 5;
            // B fragment: lane = column n, K runs {8h..8h+7, 16+8h..16+8h+7} (fp32 -> bf16)
            v16bf bfr;
            const float* bp = &wt[buf][n * WTSTR + cc * 32 + koffB];
#pragma unroll
            for (int e = 0; e < 16; ++e) bfr[e] = (bf16)bp[e < 8 ? e : e + 8];
            // A fragment, generated
            float s; int base;
            if (MODE == 3) {
                const int i = (int)(k0 >> 12), j = (int)((k0 >> 6) & 63);
                s = frow[i] * frow[j]; base = (int)(k0 & 63);
            } else if (MODE == 2) {
                s = frow[(int)(k0 >> 6)]; base = (int)(k0 & 63);
            } else { s = 1.0f; base = (int)k0; }
            v16bf afr;
#pragma unroll
            for (int e = 0; e < 16; ++e)
                afr[e] = (bf16)(s * frow[base + koffA + (e < 8 ? e : e + 8)]);
            acc = __builtin_amdgcn_wmma_f32_16x16x32_bf16(
                false, afr, false, bfr, (short)0, acc, false, false);
        }
    };

#if POLY_ASYNC
    issue(0, 0);
    issue(1, 1);
    for (int ib = 0; ib < nB; ++ib) {
        if (ib + 1 < nB) __builtin_amdgcn_s_wait_asynccnt(CB);
        else             __builtin_amdgcn_s_wait_asynccnt(0);
        __syncthreads();                 // stage ib visible to all waves (+fcs on ib==0)
        compute(ib, ib & 1);
        __syncthreads();                 // everyone done reading buf before re-fill
        issue(ib + 2, ib & 1);
    }
#else
    for (int ib = 0; ib < nB; ++ib) {
        __syncthreads();
        issue(ib, 0);
        __syncthreads();
        compute(ib, 0);
    }
#endif

    // C layout: VGPR v, lanes 0-15 -> M=v, N=lane; lanes 16-31 -> M=v+8
    const int slot = slotBase + split;
#pragma unroll
    for (int v = 0; v < 8; ++v) {
        const int M = v + ((lane >> 4) << 3);
        const int bb = (wv << 4) + M;
        const int o  = obase + (lane & 15);
        partial[((long long)slot * 64 + bb) * 1024 + o] = acc[v];
    }
}

// ======================================================================
// K5: out = bm + bc + b3 + sum over 21 partial slots
// ======================================================================
__global__ __launch_bounds__(256) void finalize_kernel(
    const float* __restrict__ partial, const float* __restrict__ bm,
    const float* __restrict__ bc, const float* __restrict__ b3,
    float* __restrict__ out)
{
    const int idx = blockIdx.x * 256 + threadIdx.x;
    if (idx >= 64 * 1024) return;
    const int o = idx & 1023;
    float s = bm[o] + bc[o] + b3[o];
#pragma unroll
    for (int slot = 0; slot < 21; ++slot)
        s += partial[slot * 65536 + idx];
    out[idx] = s;
}

// ======================================================================
// Launch
// ======================================================================
extern "C" void kernel_launch(void* const* d_in, const int* in_sizes, int n_in,
                              void* d_out, int out_size, void* d_ws, size_t ws_size,
                              hipStream_t stream)
{
    const float* x  = (const float*)d_in[0];
    const float* w1 = (const float*)d_in[1];
    const float* b1 = (const float*)d_in[2];
    const float* w2 = (const float*)d_in[3];
    const float* b2 = (const float*)d_in[4];
    const float* wm = (const float*)d_in[5];
    const float* bm = (const float*)d_in[6];
    const float* wc = (const float*)d_in[7];
    const float* bc = (const float*)d_in[8];
    const float* w3 = (const float*)d_in[9];
    const float* b3 = (const float*)d_in[10];

    char* ws = (char*)d_ws;
    bf16*  h1b      = (bf16*) (ws);                    // 51,380,224 B
    float* featpart = (float*)(ws + 51380224);         //    802,816 B (64*49*64)
    float* feat     = (float*)(ws + 52183040);         //     16,384 B
    float* fc       = (float*)(ws + 52199424);         //     16,384 B
    float* partial  = (float*)(ws + 52215808);         //  5,505,024 B (21 slots)

    conv1_kernel<<<dim3(49, 32, 64), 256, 0, stream>>>(x, w1, b1, h1b);
    conv2_pool_kernel<<<dim3(49, 64), 128, 0, stream>>>(h1b, w2, b2, featpart);
    feat_fc_kernel<<<64, 64, 0, stream>>>(featpart, feat, fc);

    // slots: MODE3 -> 0..15, MODE2 -> 16..19, MODE1 -> 20
    poly_gemm_kernel<3><<<dim3(64, 16), 128, 0, stream>>>(w3, fc,   partial, 512, 0);
    poly_gemm_kernel<2><<<dim3(64, 4),  128, 0, stream>>>(wc, fc,   partial, 32, 16);
    poly_gemm_kernel<1><<<dim3(64, 1),  128, 0, stream>>>(wm, feat, partial, 2,  20);

    finalize_kernel<<<256, 256, 0, stream>>>(partial, bm, bc, b3, (float*)d_out);
}
